// CfCCell_48619029791332
// MI455X (gfx1250) — matrix operand order repaired and checked by reading, not compiled, not run on hardware
//
#include <hip/hip_runtime.h>
#include <hip/hip_bf16.h>

// CfC cell on MI455X (gfx1250, wave32, WMMA).
// pack f32->f16 -> GEMM1(lecun_tanh) -> GEMM2(lecun_tanh) -> fused 4-head GEMM + CfC gate.
// Matmuls: v_wmma_f32_16x16x32_f16. Backbone GEMM stages tiles via async global->LDS
// (ASYNCcnt double-buffering) when the toolchain exposes the builtin.

typedef __attribute__((ext_vector_type(16))) _Float16 v16h;
typedef __attribute__((ext_vector_type(8)))  _Float16 v8h;
typedef __attribute__((ext_vector_type(8)))  float    v8f;

// async-to-LDS builtin uses GCC-vector int4 pointers with explicit address spaces
typedef int b128_t __attribute__((vector_size(16)));
typedef __attribute__((address_space(1))) b128_t* as1_b128p;
typedef __attribute__((address_space(3))) b128_t* as3_b128p;

#define B_ROWS  8192
#define IN_DIM  512
#define HID_DIM 512
#define BB_DIM  1024
#define K1      1024   // IN + HID

#if __has_builtin(__builtin_amdgcn_global_load_async_to_lds_b128)
#define HAVE_ASYNC_LDS 1
#else
#define HAVE_ASYNC_LDS 0
#endif

#if __has_builtin(__builtin_amdgcn_s_wait_asynccnt)
#define WAIT_ASYNC(n) __builtin_amdgcn_s_wait_asynccnt(n)
#else
#define WAIT_ASYNC(n) asm volatile("s_wait_asynccnt " #n ::: "memory")
#endif

#define LDS_STRIDE 40   // halfs per staged row (32 data + 8 pad) -> conflict-free frag reads

// ---- fast branchless activations ----
__device__ __forceinline__ float fast_tanh(float x) {
#if __has_builtin(__builtin_amdgcn_tanhf)
  return __builtin_amdgcn_tanhf(x);
#elif __has_builtin(__builtin_amdgcn_tanh_f32)
  return __builtin_amdgcn_tanh_f32(x);
#else
  float ax = __builtin_fabsf(x);
  float e  = __builtin_amdgcn_exp2f(ax * -2.88539008177793f);
  float r  = (1.0f - e) * __builtin_amdgcn_rcpf(1.0f + e);
  return __builtin_copysignf(r, x);
#endif
}

__device__ __forceinline__ float fast_sigmoid(float z) {
  float e = __builtin_amdgcn_exp2f(z * -1.44269504088896f);
  return __builtin_amdgcn_rcpf(1.0f + e);
}

// ---- async global(16B/lane) -> LDS ----
__device__ __forceinline__ void async_g2l_b128(const _Float16* g, _Float16* l) {
#if HAVE_ASYNC_LDS
  __builtin_amdgcn_global_load_async_to_lds_b128(
      (as1_b128p)(void*)(g), (as3_b128p)(void*)(l), 0, 0);
#endif
}

// stage one 128x32 f16 tile (rows stride ld in global) into LDS (row stride LDS_STRIDE)
// 512 x 16B transfers, 4 per thread (128-thread block)
__device__ __forceinline__ void stage_tile_async(const _Float16* __restrict__ g, int ld,
                                                 _Float16* l, int tid) {
#pragma unroll
  for (int c = 0; c < 4; ++c) {
    int u   = tid + 128 * c;        // 0..511
    int row = u >> 2;
    int col = (u & 3) * 8;          // halfs
    async_g2l_b128(g + (size_t)row * ld + col, l + row * LDS_STRIDE + col);
  }
}

// ---- WMMA fragment loaders (wave32 layouts per CDNA5 ISA §7.12.2) ----
// A 16x32 f16: lane L -> row L%16; halves 0..7 = K kb..kb+7, halves 8..15 = kb+16..kb+23
__device__ __forceinline__ v16h load_a_frag(const _Float16* __restrict__ t, int ld, int lane) {
  const int r  = lane & 15;
  const int kb = (lane >> 4) << 3;
  const _Float16* p = t + (size_t)r * ld + kb;
  union { v16h v; float4 q[2]; } u;
  u.q[0] = *(const float4*)(p);
  u.q[1] = *(const float4*)(p + 16);
  return u.v;
}

// B 32x16 f16 (W row n feeds B column n): lane L -> col L%16; halves 0..7 = K kb..kb+7, halves 8..15 = kb+16..kb+23
__device__ __forceinline__ v16h load_b_frag(const _Float16* __restrict__ t, int ld, int lane) {
  const int n  = lane & 15;
  const int kb = (lane >> 4) << 3;
  const _Float16* p = t + (size_t)n * ld + kb;
  union { v16h v; float4 q[2]; } u;
  u.q[0] = *(const float4*)(p);
  u.q[1] = *(const float4*)(p + 16);
  return u.v;
}

// ---- pack kernels: f32 -> f16 row-major ----
__global__ void pack_concat(const float* __restrict__ in, const float* __restrict__ hx,
                            _Float16* __restrict__ X) {
  size_t t = (size_t)blockIdx.x * blockDim.x + threadIdx.x;
  size_t base = t * 8;
  if (base >= (size_t)B_ROWS * K1) return;
  int m = (int)(base >> 10);
  int k = (int)(base & (K1 - 1));
  const float* s = (k < IN_DIM) ? (in + (size_t)m * IN_DIM + k)
                                : (hx + (size_t)m * HID_DIM + (k - IN_DIM));
  float4 f0 = *(const float4*)(s);
  float4 f1 = *(const float4*)(s + 4);
  v8h o;
  o[0] = (_Float16)f0.x; o[1] = (_Float16)f0.y; o[2] = (_Float16)f0.z; o[3] = (_Float16)f0.w;
  o[4] = (_Float16)f1.x; o[5] = (_Float16)f1.y; o[6] = (_Float16)f1.z; o[7] = (_Float16)f1.w;
  *(v8h*)(X + base) = o;
}

__global__ void pack_f16(const float* __restrict__ src, _Float16* __restrict__ dst, int total8) {
  int t = blockIdx.x * blockDim.x + threadIdx.x;
  if (t >= total8) return;
  const float* s = src + (size_t)t * 8;
  float4 f0 = *(const float4*)(s);
  float4 f1 = *(const float4*)(s + 4);
  v8h o;
  o[0] = (_Float16)f0.x; o[1] = (_Float16)f0.y; o[2] = (_Float16)f0.z; o[3] = (_Float16)f0.w;
  o[4] = (_Float16)f1.x; o[5] = (_Float16)f1.y; o[6] = (_Float16)f1.z; o[7] = (_Float16)f1.w;
  *(v8h*)(dst + (size_t)t * 8) = o;
}

// ---- backbone GEMM: Y = lecun_tanh(A @ W^T + bias), f16 out ----
// block = 128 thr (4 waves as 2x2), block tile 128x128, wave tile 64x64 (16 WMMAs / K-step)
__global__ __launch_bounds__(128) void gemm_act(
    const _Float16* __restrict__ A,   // [M][K]
    const _Float16* __restrict__ W,   // [N][K]
    const float*    __restrict__ bias,// [N]
    _Float16*       __restrict__ Y,   // [M][N]
    int M, int N, int K) {
  const int tid  = threadIdx.x;
  const int lane = tid & 31;
  const int w    = tid >> 5;
  const int wm   = w >> 1;
  const int wn   = w & 1;
  const int mblock = blockIdx.x * 128;
  const int nblock = blockIdx.y * 128;
  const int mbase = mblock + wm * 64;
  const int nbase = nblock + wn * 64;

  v8f acc[4][4] = {};

#if HAVE_ASYNC_LDS
  __shared__ _Float16 sA[2][128 * LDS_STRIDE];
  __shared__ _Float16 sB[2][128 * LDS_STRIDE];
  // prefetch K-tile 0
  stage_tile_async(A + (size_t)mblock * K, K, sA[0], tid);
  stage_tile_async(W + (size_t)nblock * K, K, sB[0], tid);
  int buf = 0;
  for (int kt = 0; kt < K; kt += 32, buf ^= 1) {
    if (kt + 32 < K) {
      stage_tile_async(A + (size_t)mblock * K + (kt + 32), K, sA[buf ^ 1], tid);
      stage_tile_async(W + (size_t)nblock * K + (kt + 32), K, sB[buf ^ 1], tid);
      WAIT_ASYNC(8);    // 8 in flight for kt+32 -> tile kt fully landed (in-order per wave)
    } else {
      WAIT_ASYNC(0);
    }
    __syncthreads();
    v16h a[4], b[4];
#pragma unroll
    for (int i = 0; i < 4; ++i)
      a[i] = load_a_frag(sA[buf] + (wm * 64 + i * 16) * LDS_STRIDE, LDS_STRIDE, lane);
#pragma unroll
    for (int j = 0; j < 4; ++j)
      b[j] = load_b_frag(sB[buf] + (wn * 64 + j * 16) * LDS_STRIDE, LDS_STRIDE, lane);
#pragma unroll
    for (int i = 0; i < 4; ++i)
#pragma unroll
      for (int j = 0; j < 4; ++j)
        acc[i][j] = __builtin_amdgcn_wmma_f32_16x16x32_f16(
            false, a[i], false, b[j], (short)0, acc[i][j], false, false);
    __syncthreads();    // all waves done reading buf before next iter's async overwrites it
  }
#else
  for (int kt = 0; kt < K; kt += 32) {
    v16h a[4], b[4];
#pragma unroll
    for (int i = 0; i < 4; ++i)
      a[i] = load_a_frag(A + (size_t)(mbase + 16 * i) * K + kt, K, lane);
#pragma unroll
    for (int j = 0; j < 4; ++j)
      b[j] = load_b_frag(W + (size_t)(nbase + 16 * j) * K + kt, K, lane);
#pragma unroll
    for (int i = 0; i < 4; ++i)
#pragma unroll
      for (int j = 0; j < 4; ++j)
        acc[i][j] = __builtin_amdgcn_wmma_f32_16x16x32_f16(
            false, a[i], false, b[j], (short)0, acc[i][j], false, false);
  }
#endif

  // C/D layout: VGPR p, lane L -> C[p + 8*(L/16)][L%16]
  const int nlo = lane & 15;
  const int mo  = (lane >> 4) * 8;
#pragma unroll
  for (int j = 0; j < 4; ++j) {
    const int n = nbase + 16 * j + nlo;
    const float bn = bias[n];
#pragma unroll
    for (int i = 0; i < 4; ++i) {
#pragma unroll
      for (int p = 0; p < 8; ++p) {
        float y = 1.7159f * fast_tanh(0.666f * (acc[i][j][p] + bn));
        int m = mbase + 16 * i + mo + p;
        Y[(size_t)m * N + n] = (_Float16)y;
      }
    }
  }
}

// ---- fused head kernel: 4 GEMMs sharing A, then CfC gate, fp32 out ----
// block = 256 thr (8 waves as 2x4), block tile 128M x 64N, wave tile 64M x 16N x 4 matrices
__global__ __launch_bounds__(256) void heads_fused(
    const _Float16* __restrict__ A,    // [M][K] = backbone output
    const _Float16* __restrict__ W4,   // [4][HID][K]: ff1, ff2, ta, tb
    const float* __restrict__ bff1, const float* __restrict__ bff2,
    const float* __restrict__ bta,  const float* __restrict__ btb,
    const float* __restrict__ ts,      // [M]
    float* __restrict__ out,           // [M][HID]
    int M, int K) {
  const int lane = threadIdx.x & 31;
  const int w    = threadIdx.x >> 5;
  const int wm   = w >> 2;                  // 0..1
  const int wn   = w & 3;                   // 0..3
  const int mbase = blockIdx.x * 128 + wm * 64;
  const int nbase = blockIdx.y * 64 + wn * 16;

  v8f acc[4][4] = {};                       // [mtile][matrix]
  for (int kt = 0; kt < K; kt += 32) {
    v16h a[4], b[4];
#pragma unroll
    for (int i = 0; i < 4; ++i)
      a[i] = load_a_frag(A + (size_t)(mbase + 16 * i) * K + kt, K, lane);
#pragma unroll
    for (int q = 0; q < 4; ++q)
      b[q] = load_b_frag(W4 + ((size_t)q * HID_DIM + nbase) * K + kt, K, lane);
#pragma unroll
    for (int i = 0; i < 4; ++i)
#pragma unroll
      for (int q = 0; q < 4; ++q)
        acc[i][q] = __builtin_amdgcn_wmma_f32_16x16x32_f16(
            false, a[i], false, b[q], (short)0, acc[i][q], false, false);
  }

  const int n  = nbase + (lane & 15);
  const int mo = (lane >> 4) * 8;
  const float b1 = bff1[n], b2 = bff2[n], b3 = bta[n], b4 = btb[n];
#pragma unroll
  for (int i = 0; i < 4; ++i) {
#pragma unroll
    for (int p = 0; p < 8; ++p) {
      const int m = mbase + 16 * i + mo + p;
      const float tsv = ts[m];
      const float f1 = fast_tanh(acc[i][0][p] + b1);
      const float f2 = fast_tanh(acc[i][1][p] + b2);
      const float ta = acc[i][2][p] + b3;
      const float tb = acc[i][3][p] + b4;
      const float s  = fast_sigmoid(ta * tsv + tb);
      out[(size_t)m * HID_DIM + n] = f1 + s * (f2 - f1);
    }
  }
}

extern "C" void kernel_launch(void* const* d_in, const int* in_sizes, int n_in,
                              void* d_out, int out_size, void* d_ws, size_t ws_size,
                              hipStream_t stream) {
  const float* input = (const float*)d_in[0];
  const float* hx    = (const float*)d_in[1];
  const float* ts    = (const float*)d_in[2];
  const float* Wb1   = (const float*)d_in[3];
  const float* bb1   = (const float*)d_in[4];
  const float* Wb2   = (const float*)d_in[5];
  const float* bb2   = (const float*)d_in[6];
  const float* Wff1  = (const float*)d_in[7];
  const float* bff1  = (const float*)d_in[8];
  const float* Wff2  = (const float*)d_in[9];
  const float* bff2  = (const float*)d_in[10];
  const float* Wta   = (const float*)d_in[11];
  const float* bta   = (const float*)d_in[12];
  const float* Wtb   = (const float*)d_in[13];
  const float* btb   = (const float*)d_in[14];

  char* ws = (char*)d_ws;
  const size_t MB = 1024u * 1024u;
  _Float16* Xh  = (_Float16*)(ws + 0);         // 16 MB (reused for Y2)
  _Float16* W1h = (_Float16*)(ws + 16 * MB);   //  2 MB
  _Float16* W2h = (_Float16*)(ws + 18 * MB);   //  2 MB
  _Float16* W4h = (_Float16*)(ws + 20 * MB);   //  4 MB: [4][512][1024]
  _Float16* Y1  = (_Float16*)(ws + 24 * MB);   // 16 MB
  _Float16* Y2  = Xh;

  pack_concat<<<4096, 256, 0, stream>>>(input, hx, Xh);
  pack_f16<<<512, 256, 0, stream>>>(Wb1, W1h, (BB_DIM * K1) / 8);
  pack_f16<<<512, 256, 0, stream>>>(Wb2, W2h, (BB_DIM * BB_DIM) / 8);
  pack_f16<<<256, 256, 0, stream>>>(Wff1, W4h + (size_t)0 * HID_DIM * BB_DIM, (HID_DIM * BB_DIM) / 8);
  pack_f16<<<256, 256, 0, stream>>>(Wff2, W4h + (size_t)1 * HID_DIM * BB_DIM, (HID_DIM * BB_DIM) / 8);
  pack_f16<<<256, 256, 0, stream>>>(Wta,  W4h + (size_t)2 * HID_DIM * BB_DIM, (HID_DIM * BB_DIM) / 8);
  pack_f16<<<256, 256, 0, stream>>>(Wtb,  W4h + (size_t)3 * HID_DIM * BB_DIM, (HID_DIM * BB_DIM) / 8);

  gemm_act<<<dim3(B_ROWS / 128, BB_DIM / 128), 128, 0, stream>>>(Xh, W1h, bb1, Y1, B_ROWS, BB_DIM, K1);
  gemm_act<<<dim3(B_ROWS / 128, BB_DIM / 128), 128, 0, stream>>>(Y1, W2h, bb2, Y2, B_ROWS, BB_DIM, BB_DIM);

  heads_fused<<<dim3(B_ROWS / 128, HID_DIM / 64), 256, 0, stream>>>(
      Y2, W4h, bff1, bff2, bta, btb, ts, (float*)d_out, B_ROWS, BB_DIM);
}
